// RNN_23356032155904
// MI455X (gfx1250) — matrix operand order repaired and checked by reading, not compiled, not run on hardware
//
#include <hip/hip_runtime.h>
#include <hip/hip_bf16.h>

typedef __attribute__((ext_vector_type(16))) __bf16 v16bf;
typedef __attribute__((ext_vector_type(8)))  __bf16 v8bf;
typedef __attribute__((ext_vector_type(2)))  __bf16 v2bf;
typedef __attribute__((ext_vector_type(8)))  float  v8f;

#define B_TOT 256
#define T_LEN 2048
#define DIN   128
#define HDIM  128
#define G4    512
#define BTILE 16
#define NTHREADS 1024

// Branch-free activations: only v_exp_f32 (+mul/add), no ocml large-arg branches.
__device__ __forceinline__ float fast_sigmoid(float x) {
    return 1.0f / (1.0f + __expf(-x));
}
__device__ __forceinline__ float fast_tanh(float x) {
    return 2.0f / (1.0f + __expf(-2.0f * x)) - 1.0f;
}

// A-fragment (16x32 bf16, M x K) from a row-major [16 x 128] bf16 LDS tile.
// Layout (ISA 7.12.2, 16-bit A 16x32): lanes 0-15 hold M=lane, K = k0+{0..7,16..23};
// lanes 16-31 hold M=lane-16, K = k0+{8..15,24..31}.
__device__ __forceinline__ v16bf load_a_frag(const __bf16* lds, int m, int k0, int lane) {
    const int base = k0 + ((lane & 16) ? 8 : 0);
    const __bf16* p = lds + m * 128 + base;
    union { v16bf v; struct { v8bf lo, hi; } s; } u;
    u.s.lo = *(const v8bf*)(p);        // K = base .. base+7
    u.s.hi = *(const v8bf*)(p + 16);   // K = base+16 .. base+23
    return u.v;
}

// B-fragment (32x16 bf16, K x N) for W^T where W is row-major [N_total x 128] f32.
// W^T(k, n) = W[n, k] -> lane (N = col) reads 16 K (two runs of 8, 16 apart) from row `col`.
// Lanes 0-15: N=lane, K = k0+{0..7,16..23} ; lanes 16-31: N=lane-16, K = k0+{8..15,24..31}.
__device__ __forceinline__ v16bf load_b_frag(const float* __restrict__ W, int col, int k0, int lane) {
    const int kb = k0 + ((lane & 16) ? 8 : 0);
    const float* src = W + col * 128 + kb;
    v16bf b;
#pragma unroll
    for (int i = 0; i < 16; ++i) b[i] = (__bf16)src[i < 8 ? i : i + 8];
    return b;
}

__global__ __launch_bounds__(NTHREADS)
void lstm_persistent_kernel(const float* __restrict__ x,
                            const float* __restrict__ Wih,
                            const float* __restrict__ Whh,
                            const float* __restrict__ bih,
                            const float* __restrict__ bhh,
                            const float* __restrict__ Wo,
                            const float* __restrict__ bo,
                            float* __restrict__ out) {
    __shared__ __align__(16) float   gates[BTILE * G4];   // 32 KB, fp32 gate staging
    __shared__ __align__(16) __bf16  xs[BTILE * DIN];     // 4 KB, x_t tile bf16
    __shared__ __align__(16) __bf16  hs[BTILE * HDIM];    // 4 KB, h tile bf16

    const int tid  = threadIdx.x;
    const int lane = tid & 31;
    const int wave = tid >> 5;                // 0..31
    const int b0   = blockIdx.x * BTILE;      // batch base for this WG

    const int n0   = wave * 16;               // gate-column tile base of this wave
    const int col  = n0 + (lane & 15);        // this lane's N column
    const int mrow = (lane & 15);             // this lane's A-matrix row

    // ---- weight-stationary: preload 8 bf16 B-fragments into VGPRs (live whole T-loop)
    v16bf bwx[4], bwh[4];
#pragma unroll
    for (int kk = 0; kk < 4; ++kk) {
        bwx[kk] = load_b_frag(Wih, col, kk * 32, lane);
        bwh[kk] = load_b_frag(Whh, col, kk * 32, lane);
    }
    const float biasv = bih[col] + bhh[col];

    // x staging mapping: 2 *consecutive* d per thread -> b64 load + packed cvt + b32 store
    const int xe  = 2 * tid;                  // element in 16x128 tile
    const int xbb = xe >> 7;                  // batch row in tile
    const int xd  = xe & 127;                 // feature index (even)
    const float* xptr = x + ((size_t)(b0 + xbb) * T_LEN) * DIN + xd;

    // init h = 0 (bf16, LDS); cell state c lives in VGPRs (2 elems / thread)
    for (int e = tid; e < BTILE * HDIM; e += NTHREADS) hs[e] = (__bf16)0.0f;
    float c0 = 0.0f, c1 = 0.0f;
    const int e0 = tid, e1 = tid + NTHREADS;  // elements of the 16x128 c/h tile
    __syncthreads();

    for (int t = 0; t < T_LEN; ++t) {
        // ---- stage x_t tile [16 x 128] f32 -> bf16 LDS: one b64 load, one b32 store
        {
            const float* px = xptr + (size_t)t * DIN;
            const float2 v = *(const float2*)px;
            v2bf pk;
            pk.x = (__bf16)v.x;
            pk.y = (__bf16)v.y;
            *(v2bf*)(&xs[xe]) = pk;
            if (t + 1 < T_LEN) __builtin_prefetch(px + DIN, 0, 3);
        }
        __syncthreads();  // xs staged; hs from previous step ready

        // ---- gates tile = bias + x_t @ Wih^T + h @ Whh^T  (8 x wmma bf16, f32 acc)
        v8f acc;
#pragma unroll
        for (int r = 0; r < 8; ++r) acc[r] = biasv;
#pragma unroll
        for (int kk = 0; kk < 4; ++kk) {
            v16bf ax = load_a_frag(xs, mrow, kk * 32, lane);
            acc = __builtin_amdgcn_wmma_f32_16x16x32_bf16(
                false, ax, false, bwx[kk], (short)0, acc, false, false);
        }
#pragma unroll
        for (int kk = 0; kk < 4; ++kk) {
            v16bf ah = load_a_frag(hs, mrow, kk * 32, lane);
            acc = __builtin_amdgcn_wmma_f32_16x16x32_bf16(
                false, ah, false, bwh[kk], (short)0, acc, false, false);
        }

        // ---- scatter f32 accumulator tile to LDS gates[16 x 512]
        {
            const int mbase = (lane & 16) ? 8 : 0;
#pragma unroll
            for (int r = 0; r < 8; ++r)
                gates[(mbase + r) * G4 + col] = acc[r];
        }
        __syncthreads();  // gates complete; all hs/xs reads done

        // ---- elementwise LSTM cell: c in VGPRs, write new h (bf16) to LDS
        {
            const int bb = e0 >> 7, j = e0 & 127;
            const float* gg = gates + bb * G4 + j;
            const float iv = fast_sigmoid(gg[0]);
            const float fv = fast_sigmoid(gg[128]);
            const float gv = fast_tanh(gg[256]);
            const float ov = fast_sigmoid(gg[384]);
            c0 = fv * c0 + iv * gv;
            hs[e0] = (__bf16)(ov * fast_tanh(c0));
        }
        {
            const int bb = e1 >> 7, j = e1 & 127;
            const float* gg = gates + bb * G4 + j;
            const float iv = fast_sigmoid(gg[0]);
            const float fv = fast_sigmoid(gg[128]);
            const float gv = fast_tanh(gg[256]);
            const float ov = fast_sigmoid(gg[384]);
            c1 = fv * c1 + iv * gv;
            hs[e1] = (__bf16)(ov * fast_tanh(c1));
        }
        // next iteration's first __syncthreads orders hs writes vs. wmma reads
    }
    __syncthreads();  // h_last finalized in hs

    // ---- output projection: out[16 x 128] = h_last @ Wo^T + bo  (waves 0..7)
    if (wave < 8) {
        const int ncol = wave * 16 + (lane & 15);
        const float bv = bo[ncol];
        v8f acc;
#pragma unroll
        for (int r = 0; r < 8; ++r) acc[r] = bv;
#pragma unroll
        for (int kk = 0; kk < 4; ++kk) {
            v16bf ah = load_a_frag(hs, mrow, kk * 32, lane);
            v16bf bw = load_b_frag(Wo, ncol, kk * 32, lane);
            acc = __builtin_amdgcn_wmma_f32_16x16x32_bf16(
                false, ah, false, bw, (short)0, acc, false, false);
        }
        const int mbase = (lane & 16) ? 8 : 0;
#pragma unroll
        for (int r = 0; r < 8; ++r)
            out[(size_t)(b0 + mbase + r) * 128 + ncol] = acc[r];
    }
}

extern "C" void kernel_launch(void* const* d_in, const int* in_sizes, int n_in,
                              void* d_out, int out_size, void* d_ws, size_t ws_size,
                              hipStream_t stream) {
    const float* x   = (const float*)d_in[0];  // [256, 2048, 128]
    const float* Wih = (const float*)d_in[1];  // [512, 128]
    const float* Whh = (const float*)d_in[2];  // [512, 128]
    const float* bih = (const float*)d_in[3];  // [512]
    const float* bhh = (const float*)d_in[4];  // [512]
    const float* Wo  = (const float*)d_in[5];  // [128, 128]
    const float* bo  = (const float*)d_in[6];  // [128]
    float* out = (float*)d_out;                // [256, 128]

    (void)in_sizes; (void)n_in; (void)out_size; (void)d_ws; (void)ws_size;

    dim3 grid(B_TOT / BTILE);   // 16 persistent workgroups, batch-partitioned
    dim3 block(NTHREADS);       // 32 waves: one 16x16 gate tile per wave
    hipLaunchKernelGGL(lstm_persistent_kernel, grid, block, 0, stream,
                       x, Wih, Whh, bih, bhh, Wo, bo, out);
}
